// Repro_17282948399380
// MI455X (gfx1250) — matrix operand order repaired and checked by reading, not compiled, not run on hardware
//
#include <hip/hip_runtime.h>
#include <stdint.h>
#include <stddef.h>

// ---------------- constants ----------------
constexpr int kB  = 256;
constexpr int kS  = 128;
constexpr int kD  = 256;
constexpr int kH  = 4;
constexpr int kHD = 64;
constexpr int kF  = 2048;
constexpr int kL  = 6;
constexpr int kBS = kB * kS;                       // 32768 rows
constexpr float kNeg = -3.4028234663852886e38f;

typedef __bf16 bf16;
typedef __attribute__((ext_vector_type(16))) __bf16 v16bf;
typedef __attribute__((ext_vector_type(8)))  __bf16 v8bf;
typedef __attribute__((ext_vector_type(8)))  float  v8f;

// ---------------- device helpers ----------------
__device__ __forceinline__ bf16 f2bf(float f) {
  union { float f; uint32_t u; } c; c.f = f;
  uint32_t r = c.u + 0x7FFFu + ((c.u >> 16) & 1u);   // round-to-nearest-even
  union { uint16_t s; bf16 b; } o; o.s = (uint16_t)(r >> 16);
  return o.b;
}

__device__ __forceinline__ v8f vzero8() {
  v8f v;
#pragma unroll
  for (int i = 0; i < 8; ++i) v[i] = 0.f;
  return v;
}

__device__ __forceinline__ v8f wmma_bf16(v16bf a, v16bf b, v8f c) {
  return __builtin_amdgcn_wmma_f32_16x16x32_bf16(false, a, false, b, (short)0, c,
                                                 false, false);
}

// A tile 16x32 (MxK) from row-major source, leading dim ld.
// lanes 0-15: row=lane, K = kb+0..15; lanes 16-31: row=lane-16, K = kb+16..31
__device__ __forceinline__ v16bf load_a_tile(const bf16* base, int ld, int row_base,
                                             int kb, int lane) {
  const int r  = lane & 15;
  const int hi = lane >> 4;
  const bf16* p = base + (size_t)(row_base + r) * ld + kb + hi * 16;
  v8bf lo = *reinterpret_cast<const v8bf*>(p);
  v8bf hh = *reinterpret_cast<const v8bf*>(p + 8);
  v16bf out;
#pragma unroll
  for (int i = 0; i < 8; ++i) { out[i] = lo[i]; out[8 + i] = hh[i]; }
  return out;
}

// B tile 32x16 (KxN) for logical B = W^T where W is row-major [N, ld].
// lane holds column n=lane&15 (= W row n); lanes 0-15 K=kb..kb+15, lanes 16-31 K=kb+16..kb+31
__device__ __forceinline__ v16bf load_bt_tile(const bf16* Wm, int ld, int n_base,
                                              int kb, int lane) {
  const int n  = lane & 15;
  const int hi = lane >> 4;
  const bf16* p = Wm + (size_t)(n_base + n) * ld + kb + hi * 16;
  v8bf lo = *reinterpret_cast<const v8bf*>(p);
  v8bf hh = *reinterpret_cast<const v8bf*>(p + 8);
  v16bf out;
#pragma unroll
  for (int i = 0; i < 8; ++i) { out[i] = lo[i]; out[8 + i] = hh[i]; }
  return out;
}

// B tile 32x16 (KxN) from a row-major [K, ld] (non-transposed) source.
__device__ __forceinline__ v16bf load_bn_tile(const bf16* Vm, int ld, int n_base,
                                              int kb, int lane) {
  const int n  = lane & 15;
  const int hi = lane >> 4;
  const bf16* p = Vm + (size_t)(kb + hi * 16) * ld + n_base + n;
  v16bf out;
#pragma unroll
  for (int j = 0; j < 16; ++j) out[j] = p[(size_t)j * ld];
  return out;
}

// C-layout epilogue store: VGPR j -> row row0+j, col fixed per lane.
__device__ __forceinline__ void epi_store(v8f acc, int row0, int col, int ldc,
                                          const float* bias, float scale, int relu,
                                          float* outF, bf16* outB) {
  const float bv = bias[col];
#pragma unroll
  for (int j = 0; j < 8; ++j) {
    float v = (acc[j] + bv) * scale;
    if (relu) v = fmaxf(v, 0.f);
    const size_t idx = (size_t)(row0 + j) * ldc + col;
    if (outF) outF[idx] = v;
    if (outB) outB[idx] = f2bf(v);
  }
}

// ---------------- GEMM: C[M,N] = act(scale * (A[M,K] * W[N,K]^T + bias)) ----------------
// Block = 4 waves; each wave computes 32x64; block tile = 32(M) x 256(N).
// The shared 32-row A panel is staged into LDS with async copies (ASYNCcnt),
// double-buffered so the copy of panel k+1 overlaps WMMA on panel k.
// Grid: (M/32, N/256). All dims are exact multiples here (no guards: loop has barriers).
__global__ __launch_bounds__(128) void gemm_xwt(const bf16* __restrict__ A,
                                                const bf16* __restrict__ Wm,
                                                const float* __restrict__ bias,
                                                float* outF, bf16* outB,
                                                int M, int N, int K,
                                                float scale, int relu) {
  __shared__ bf16 aPanel[2][32][32];                 // 2 x 2KB double buffer
  const int lane = threadIdx.x & 31;
  const int wid  = threadIdx.x >> 5;
  const int m0 = blockIdx.x * 32;
  const int n0 = blockIdx.y * 256 + wid * 64;

  // async A-panel staging: 128 threads x 16B = 2KB = 32 rows x 32 bf16
  const int arow = threadIdx.x >> 2;                 // 0..31
  const int acol = (threadIdx.x & 3) * 8;            // element offset: 0,8,16,24
  const bf16* agBase = A + (size_t)(m0 + arow) * K + acol;
  const uint32_t ldsOff[2] = {
      (uint32_t)(uintptr_t)&aPanel[0][arow][acol],   // flat LDS addr low 32b = DS offset
      (uint32_t)(uintptr_t)&aPanel[1][arow][acol] };

  {  // prime buffer 0 with the k=0 panel
    const uint64_t ga = (uint64_t)(uintptr_t)agBase;
    asm volatile("global_load_async_to_lds_b128 %0, %1, off"
                 :: "v"(ldsOff[0]), "v"(ga) : "memory");
  }

  v8f acc[2][4];
#pragma unroll
  for (int mi = 0; mi < 2; ++mi)
#pragma unroll
    for (int ni = 0; ni < 4; ++ni) acc[mi][ni] = vzero8();

  int buf = 0;
  for (int kb = 0; kb < K; kb += 32) {
    if (kb + 32 < K) {
      const uint64_t ga = (uint64_t)(uintptr_t)(agBase + kb + 32);
      asm volatile("global_load_async_to_lds_b128 %0, %1, off"
                   :: "v"(ldsOff[buf ^ 1]), "v"(ga) : "memory");
      // async loads complete in order: <=1 outstanding => current panel landed
      asm volatile("s_wait_asynccnt 0x1" ::: "memory");
      __builtin_prefetch(Wm + (size_t)(n0 + (lane & 15)) * K + kb + 32, 0, 1);
    } else {
      asm volatile("s_wait_asynccnt 0x0" ::: "memory");
    }
    __syncthreads();                                 // panel visible to all 4 waves

    v16bf aT0 = load_a_tile(&aPanel[buf][0][0], 32, 0,  0, lane);
    v16bf aT1 = load_a_tile(&aPanel[buf][0][0], 32, 16, 0, lane);
#pragma unroll
    for (int ni = 0; ni < 4; ++ni) {
      v16bf bT = load_bt_tile(Wm, K, n0 + ni * 16, kb, lane);
      acc[0][ni] = wmma_bf16(aT0, bT, acc[0][ni]);
      acc[1][ni] = wmma_bf16(aT1, bT, acc[1][ni]);
    }
    __syncthreads();                                 // done reading before overwrite
    buf ^= 1;
  }

  const int n  = lane & 15;
  const int hi = lane >> 4;
#pragma unroll
  for (int mi = 0; mi < 2; ++mi)
#pragma unroll
    for (int ni = 0; ni < 4; ++ni)
      epi_store(acc[mi][ni], m0 + mi * 16 + hi * 8, n0 + ni * 16 + n, N,
                bias, scale, relu, outF, outB);
}

// ---------------- fused causal attention ----------------
// One wave handles (b, h, 16 query rows): 16 WMMA for scores, register softmax
// with shfl reductions, LDS restage (C-layout -> A-layout), 16 WMMA for P*V.
__global__ __launch_bounds__(128) void attn_kernel(const bf16* __restrict__ Q,
                                                   const bf16* __restrict__ Kt,
                                                   const bf16* __restrict__ V,
                                                   bf16* __restrict__ ctx) {
  __shared__ bf16 plds[4][16][kS];   // 16 KB: per-wave probability staging
  const int lane = threadIdx.x & 31;
  const int wid  = threadIdx.x >> 5;
  const int wg   = blockIdx.x * 4 + wid;
  const int qt = wg & 7;            // query tile (S/16 = 8)
  const int h  = (wg >> 3) & 3;     // head
  const int b  = wg >> 5;           // batch
  const int q0 = qt * 16;
  const int n  = lane & 15;
  const int hi = lane >> 4;

  const bf16* Qb = Q  + (size_t)b * kS * kD + h * kHD;
  const bf16* Kb = Kt + (size_t)b * kS * kD + h * kHD;
  const bf16* Vb = V  + (size_t)b * kS * kD + h * kHD;

  // scores = Q * K^T over HD=64 (2 k-steps); K stored [S, D] row-major -> transposed-B path
  v16bf aq0 = load_a_tile(Qb, kD, q0, 0,  lane);
  v16bf aq1 = load_a_tile(Qb, kD, q0, 32, lane);
  v8f sc[8];
#pragma unroll
  for (int t = 0; t < 8; ++t) {
    v8f acc = vzero8();
    acc = wmma_bf16(aq0, load_bt_tile(Kb, kD, t * 16, 0,  lane), acc);
    acc = wmma_bf16(aq1, load_bt_tile(Kb, kD, t * 16, 32, lane), acc);
    sc[t] = acc;
  }

  // causal mask + softmax. Row r = q0 + hi*8 + j lives in one 16-lane half:
  // its 128 cols are (tile t, lane n) -> exact row reduction = local-over-t + shfl_xor {1,2,4,8}.
#pragma unroll
  for (int j = 0; j < 8; ++j) {
    const int row = q0 + hi * 8 + j;
    float m = kNeg;
#pragma unroll
    for (int t = 0; t < 8; ++t) {
      const int col = t * 16 + n;
      float v = sc[t][j];
      if (col > row) v = kNeg;
      sc[t][j] = v;
      m = fmaxf(m, v);
    }
#pragma unroll
    for (int mk = 8; mk >= 1; mk >>= 1) m = fmaxf(m, __shfl_xor(m, mk, 32));
    float s = 0.f;
#pragma unroll
    for (int t = 0; t < 8; ++t) {
      const float e = __expf(sc[t][j] - m);
      sc[t][j] = e;
      s += e;
    }
#pragma unroll
    for (int mk = 8; mk >= 1; mk >>= 1) s += __shfl_xor(s, mk, 32);
    const float inv = 1.f / s;
#pragma unroll
    for (int t = 0; t < 8; ++t) sc[t][j] *= inv;
  }

  // restage probs row-major in LDS so they can be reloaded in A-matrix layout
#pragma unroll
  for (int j = 0; j < 8; ++j)
#pragma unroll
    for (int t = 0; t < 8; ++t)
      plds[wid][hi * 8 + j][t * 16 + n] = f2bf(sc[t][j]);
  __syncthreads();

  v16bf pa[4];
#pragma unroll
  for (int ks = 0; ks < 4; ++ks)
    pa[ks] = load_a_tile(&plds[wid][0][0], kS, 0, ks * 32, lane);

  // out = P[16,128] * V[128,64]; V row-major [S, D] -> non-transposed B path
#pragma unroll
  for (int nt = 0; nt < 4; ++nt) {
    v8f acc = vzero8();
#pragma unroll
    for (int ks = 0; ks < 4; ++ks)
      acc = wmma_bf16(pa[ks], load_bn_tile(Vb, kD, nt * 16, ks * 32, lane), acc);
#pragma unroll
    for (int j = 0; j < 8; ++j)
      ctx[(size_t)(b * kS + q0 + hi * 8 + j) * kD + h * kHD + nt * 16 + n] = f2bf(acc[j]);
  }
}

// ---------------- residual + LayerNorm (biased var), writes f32 state + bf16 copy ----------------
__global__ __launch_bounds__(256) void ln_kernel(const float* __restrict__ x,
                                                 const float* __restrict__ res,
                                                 const float* __restrict__ w,
                                                 const float* __restrict__ bsh,
                                                 float* __restrict__ outF,
                                                 bf16* __restrict__ outB) {
  __shared__ float red[kD];
  const int row = blockIdx.x;
  const int t = threadIdx.x;
  const size_t idx = (size_t)row * kD + t;
  const float v = x[idx] + res[idx];
  red[t] = v;
  __syncthreads();
  for (int s2 = kD / 2; s2 > 0; s2 >>= 1) { if (t < s2) red[t] += red[t + s2]; __syncthreads(); }
  const float mean = red[0] * (1.f / kD);
  __syncthreads();
  const float d = v - mean;
  red[t] = d * d;
  __syncthreads();
  for (int s2 = kD / 2; s2 > 0; s2 >>= 1) { if (t < s2) red[t] += red[t + s2]; __syncthreads(); }
  const float var = red[0] * (1.f / kD);
  const float y = d * rsqrtf(var + 1e-5f) * w[t] + bsh[t];
  if (outF) outF[idx] = y;
  if (outB) outB[idx] = f2bf(y);
}

// ---------------- embedding: OPT position ids via mask cumsum ----------------
__global__ __launch_bounds__(128) void embed_kernel(const int* __restrict__ ids,
                                                    const float* __restrict__ tok,
                                                    const float* __restrict__ pem,
                                                    float* __restrict__ xF,
                                                    bf16* __restrict__ xB) {
  __shared__ int mk[kS];
  const int b = blockIdx.x;
  const int s = threadIdx.x;
  const int id = ids[b * kS + s];
  const int m = (id != 1) ? 1 : 0;
  mk[s] = m;
  __syncthreads();
  int c = 0;
  for (int t2 = 0; t2 <= s; ++t2) c += mk[t2];
  const int p = c * m + 1;
  const float* te = tok + (size_t)id * kD;
  const float* pe = pem + (size_t)p * kD;
  const size_t base = ((size_t)b * kS + s) * kD;
  for (int d = 0; d < kD; ++d) {
    const float v = te[d] * 16.f + pe[d];
    xF[base + d] = v;
    xB[base + d] = f2bf(v);
  }
}

// ---------------- f32 -> bf16 weight cast ----------------
__global__ __launch_bounds__(256) void cvt_kernel(const float* __restrict__ in,
                                                  bf16* __restrict__ out, int nElem) {
  for (int i = blockIdx.x * 256 + threadIdx.x; i < nElem; i += gridDim.x * 256)
    out[i] = f2bf(in[i]);
}

// ---------------- host launcher ----------------
extern "C" void kernel_launch(void* const* d_in, const int* in_sizes, int n_in,
                              void* d_out, int out_size, void* d_ws, size_t ws_size,
                              hipStream_t stream) {
  (void)in_sizes; (void)n_in; (void)out_size; (void)ws_size;
  const int*   ids = (const int*)d_in[0];
  const float* tok = (const float*)d_in[1];
  const float* pem = (const float*)d_in[2];
  const float* Wq  = (const float*)d_in[3];
  const float* bq  = (const float*)d_in[4];
  const float* Wk  = (const float*)d_in[5];
  const float* bk  = (const float*)d_in[6];
  const float* Wv  = (const float*)d_in[7];
  const float* bv  = (const float*)d_in[8];
  const float* Wo  = (const float*)d_in[9];
  const float* bo  = (const float*)d_in[10];
  const float* l1w = (const float*)d_in[11];
  const float* l1b = (const float*)d_in[12];
  const float* W1  = (const float*)d_in[13];
  const float* b1  = (const float*)d_in[14];
  const float* W2  = (const float*)d_in[15];
  const float* b2  = (const float*)d_in[16];
  const float* l2w = (const float*)d_in[17];
  const float* l2b = (const float*)d_in[18];
  float* out = (float*)d_out;

  uintptr_t pp = (uintptr_t)d_ws;
  auto take = [&](size_t bytes) -> void* {
    void* r = (void*)pp;
    pp += (bytes + 255) & ~(size_t)255;
    return r;
  };
  float* xF  = (float*)take((size_t)kBS * kD * 4);   // residual stream (f32)
  float* oF  = (float*)take((size_t)kBS * kD * 4);   // block output before LN (f32)
  bf16* xBv  = (bf16*)take((size_t)kBS * kD * 2);    // residual stream (bf16 copy)
  bf16* qBv  = (bf16*)take((size_t)kBS * kD * 2);
  bf16* kBv  = (bf16*)take((size_t)kBS * kD * 2);
  bf16* vBv  = (bf16*)take((size_t)kBS * kD * 2);
  bf16* cBv  = (bf16*)take((size_t)kBS * kD * 2);    // attention context
  bf16* hBv  = (bf16*)take((size_t)kBS * kF * 2);    // FFN hidden
  bf16* WqB  = (bf16*)take((size_t)kL * kD * kD * 2);
  bf16* WkB  = (bf16*)take((size_t)kL * kD * kD * 2);
  bf16* WvB  = (bf16*)take((size_t)kL * kD * kD * 2);
  bf16* WoB  = (bf16*)take((size_t)kL * kD * kD * 2);
  bf16* W1B  = (bf16*)take((size_t)kL * kF * kD * 2);
  bf16* W2B  = (bf16*)take((size_t)kL * kD * kF * 2);

  const int nDD = kL * kD * kD;
  const int nFD = kL * kF * kD;
  cvt_kernel<<<(nDD + 255) / 256, 256, 0, stream>>>(Wq, WqB, nDD);
  cvt_kernel<<<(nDD + 255) / 256, 256, 0, stream>>>(Wk, WkB, nDD);
  cvt_kernel<<<(nDD + 255) / 256, 256, 0, stream>>>(Wv, WvB, nDD);
  cvt_kernel<<<(nDD + 255) / 256, 256, 0, stream>>>(Wo, WoB, nDD);
  cvt_kernel<<<(nFD + 255) / 256, 256, 0, stream>>>(W1, W1B, nFD);
  cvt_kernel<<<(nFD + 255) / 256, 256, 0, stream>>>(W2, W2B, nFD);

  embed_kernel<<<kB, kS, 0, stream>>>(ids, tok, pem, xF, xBv);

  auto gemm = [&](const bf16* A, const bf16* Wm, const float* bias, float* oFp,
                  bf16* oBp, int M, int N, int K, float scale, int relu) {
    dim3 g(M / 32, N / 256);
    gemm_xwt<<<g, 128, 0, stream>>>(A, Wm, bias, oFp, oBp, M, N, K, scale, relu);
  };

  for (int l = 0; l < kL; ++l) {
    const bf16* wq = WqB + (size_t)l * kD * kD;
    const bf16* wk = WkB + (size_t)l * kD * kD;
    const bf16* wv = WvB + (size_t)l * kD * kD;
    const bf16* wo = WoB + (size_t)l * kD * kD;
    const bf16* w1 = W1B + (size_t)l * kF * kD;
    const bf16* w2 = W2B + (size_t)l * kD * kF;

    gemm(xBv, wq, bq + l * kD, nullptr, qBv, kBS, kD, kD, 0.125f, 0);  // q = (xWq+b)*0.125
    gemm(xBv, wk, bk + l * kD, nullptr, kBv, kBS, kD, kD, 1.f, 0);
    gemm(xBv, wv, bv + l * kD, nullptr, vBv, kBS, kD, kD, 1.f, 0);

    attn_kernel<<<(kB * kH * (kS / 16)) / 4, 128, 0, stream>>>(qBv, kBv, vBv, cBv);

    gemm(cBv, wo, bo + l * kD, oF, nullptr, kBS, kD, kD, 1.f, 0);
    ln_kernel<<<kBS, kD, 0, stream>>>(xF, oF, l1w + l * kD, l1b + l * kD, xF, xBv);

    gemm(xBv, w1, b1 + l * kF, nullptr, hBv, kBS, kF, kD, 1.f, 1);     // ReLU
    gemm(hBv, w2, b2 + l * kD, oF, nullptr, kBS, kD, kF, 1.f, 0);

    float* lnOut = (l == kL - 1) ? out : xF;
    bf16*  lnOutB = (l == kL - 1) ? nullptr : xBv;
    ln_kernel<<<kBS, kD, 0, stream>>>(xF, oF, l2w + l * kD, l2b + l * kD, lnOut, lnOutB);
  }
}
